// DilatedMultiheadAttention_51101520888223
// MI455X (gfx1250) — matrix operand order repaired and checked by reading, not compiled, not run on hardware
//
#include <hip/hip_runtime.h>
#include <hip/hip_bf16.h>
#include <math.h>

// ---------------------------------------------------------------------------
// DilatedMultiheadAttention for MI455X (gfx1250, wave32, WMMA)
// B=1, S=4096, E=1024, H=16, d=64; configs (seg,dil) = (1024,1),(2048,2),(4096,4)
//
//  1) proj_kernel: Pq/Pk/Pv = {query,key,value} @ Wq^T (bf16 out, f32 accum,
//     v_wmma_f32_16x16x32_bf16, LDS tiled, K-step 64). Projection commutes
//     with the dilated gather, so one projection per tensor suffices.
//  2) attn_kernel: flash-style fused attention per (group, head, 128-row
//     q-block). Q/K tiles filled with GLOBAL_LOAD_ASYNC_TO_LDS_B128
//     (ASYNCcnt + s_wait_asynccnt), V transposed into LDS via VGPR staging.
//     Each wave owns 16 full score rows -> softmax reductions are 16-lane
//     shfl_xor butterflies. P staged through LDS (C->A layout).
//  3) out = sum over configs of (attn out)/3 -> fp32 atomicAdd into zeroed out.
// ---------------------------------------------------------------------------

typedef __bf16 bf16;
typedef __attribute__((ext_vector_type(16))) __bf16 v16bf;
typedef __attribute__((ext_vector_type(8)))  float  v8f;

struct alignas(16) B16 { unsigned int u[4]; };          // 16-byte POD chunk

union FragB { v16bf v; B16 q[2]; };                     // 32B bf16 operand
union FragF { v8f  v; float f[8]; };                    // 32B f32 accum

static __device__ __forceinline__ v8f wmma_bf16(const FragB& a, const FragB& b, v8f c) {
    return __builtin_amdgcn_wmma_f32_16x16x32_bf16(false, a.v, false, b.v,
                                                   (short)0, c, false, false);
}

// ---- gfx1250 async memory->LDS path (guarded; falls back to VGPR staging) --
#if defined(__has_builtin)
#if __has_builtin(__builtin_amdgcn_global_load_async_to_lds_b128)
#define HAVE_ASYNC_LDS 1
#endif
#endif

typedef int v4i __attribute__((vector_size(16)));
typedef __attribute__((address_space(1))) v4i glb_v4i;   // printed "__device__" by clang
typedef __attribute__((address_space(3))) v4i lds_v4i;   // printed "__shared__" by clang

static __device__ __forceinline__ void async_copy16(void* lds, const void* g) {
#ifdef HAVE_ASYNC_LDS
    // generic->AS pointers: AS1 shares the 64-bit VA; low 32 bits of a generic
    // shared address are the LDS byte offset (ISA aperture rules).
    __builtin_amdgcn_global_load_async_to_lds_b128(
        (glb_v4i*)(unsigned long long)(uintptr_t)g,
        (lds_v4i*)(unsigned int)(uintptr_t)lds, 0, 0);
#else
    *(B16*)lds = *(const B16*)g;
#endif
}

static __device__ __forceinline__ void wait_async0() {
#ifdef HAVE_ASYNC_LDS
#if __has_builtin(__builtin_amdgcn_s_wait_asynccnt)
    __builtin_amdgcn_s_wait_asynccnt(0);
#else
    asm volatile("s_wait_asynccnt 0" ::: "memory");
#endif
#endif
}

#define S_LEN 4096
#define E_DIM 1024
#define HDIM  64

// ===========================================================================
// Projection: out[mat][m][n] = sum_k X[m,k] * Wq[n,k]   (x @ Wq^T), bf16 out
// M total = 3*4096 (q,k,v stacked), N = 1024, K = 1024.
// Block: 256 thr (8 waves). Tile M=128, N=64, K-step 64 (2 wmma k-sub-steps).
// ===========================================================================
__global__ __launch_bounds__(256)
void proj_kernel(const float* __restrict__ xq, const float* __restrict__ xk,
                 const float* __restrict__ xv, const float* __restrict__ Wq,
                 bf16* __restrict__ out /* [3][4096][1024] bf16 */)
{
    __shared__ alignas(16) bf16 sA[128][64];  // 16 KB
    __shared__ alignas(16) bf16 sB[64][64];   //  8 KB, sB[n][k] = Wq[col0+n][k0+k]

    const int tid    = threadIdx.x;
    const int wave   = tid >> 5;
    const int lane   = tid & 31;
    const int laneHi = lane >> 4;
    const int ln     = lane & 15;

    const int row0  = blockIdx.x * 128;       // 0..12287
    const int col0  = blockIdx.y * 64;        // 0..1023
    const int mat   = row0 >> 12;             // which source matrix
    const int mrow0 = row0 & 4095;
    const float* X = (mat == 0) ? xq : (mat == 1) ? xk : xv;

    const v8f vzero = {0.f,0.f,0.f,0.f,0.f,0.f,0.f,0.f};
    v8f acc[4];
#pragma unroll
    for (int i = 0; i < 4; ++i) acc[i] = vzero;

    for (int k0 = 0; k0 < E_DIM; k0 += 64) {
        __syncthreads();
        {   // A tile 128x64: thread -> row tid/2, 32 consecutive k (f32->bf16)
            int r  = tid >> 1;
            int ks = (tid & 1) * 32;
            const float* src = X + (size_t)(mrow0 + r) * E_DIM + k0 + ks;
            bf16* dst = &sA[r][ks];
#pragma unroll
            for (int i = 0; i < 32; ++i) dst[i] = (bf16)src[i];
        }
        {   // B tile 64x64: thread -> n = tid/4, 16 consecutive k
            int n  = tid >> 2;
            int ks = (tid & 3) * 16;
            const float* src = Wq + (size_t)(col0 + n) * E_DIM + k0 + ks;
            bf16* dst = &sB[n][ks];
#pragma unroll
            for (int i = 0; i < 16; ++i) dst[i] = (bf16)src[i];
        }
        __syncthreads();

#pragma unroll
        for (int ks = 0; ks < 2; ++ks) {
            FragB a;   // A frag: row = lane&15 of wave window; k contiguous
            {
                const bf16* rp = &sA[wave * 16 + ln][ks * 32];
                int c = laneHi * 16;
                a.q[0] = *(const B16*)(rp + c);
                a.q[1] = *(const B16*)(rp + 8 + c);
            }
            FragB b[4];  // hoist all B frags -> one dscnt wait per 4 wmma
#pragma unroll
            for (int ct = 0; ct < 4; ++ct) {
                const bf16* bp = &sB[ct * 16 + ln][ks * 32 + laneHi * 16];
                b[ct].q[0] = ((const B16*)bp)[0];
                b[ct].q[1] = ((const B16*)bp)[1];
            }
#pragma unroll
            for (int ct = 0; ct < 4; ++ct)
                acc[ct] = wmma_bf16(a, b[ct], acc[ct]);
        }
    }

    // Store D: row m = v + 8*laneHi within wave window; col = ct*16 + ln
    bf16* op = out + (size_t)mat * S_LEN * E_DIM;
    const int rbase = mrow0 + wave * 16 + 8 * laneHi;
#pragma unroll
    for (int ct = 0; ct < 4; ++ct) {
        FragF c; c.v = acc[ct];
#pragma unroll
        for (int v = 0; v < 8; ++v)
            op[(size_t)(rbase + v) * E_DIM + col0 + ct * 16 + ln] = (bf16)c.f[v];
    }
}

// ===========================================================================
// Fused dilated attention. Grid: 896 blocks = 7 groups * 16 heads * 8 qblocks.
// Block: 256 thr (8 waves). Each wave: 16 q-rows. Key loop: 16 tiles of 64.
// ===========================================================================
__global__ __launch_bounds__(256)
void attn_kernel(const bf16* __restrict__ Pall /* [3][4096][1024] */,
                 const float* __restrict__ mask /* [4096] */,
                 float* __restrict__ out /* [4096][1024] f32, pre-zeroed */)
{
    const bf16* Pq = Pall;
    const bf16* Pk = Pall + (size_t)S_LEN * E_DIM;
    const bf16* Pv = Pall + (size_t)2 * S_LEN * E_DIM;

    __shared__ alignas(16) bf16 sQ[128][64];    // [qrow][dim]
    __shared__ alignas(16) bf16 sK[64][64];     // [key][dim]
    __shared__ alignas(16) bf16 sV[64][64];     // TRANSPOSED: [dim][key]
    __shared__ alignas(16) bf16 sP[8][16][64];  // per-wave probs [row][key]
    __shared__ float sM[64];

    const int b   = blockIdx.x;
    const int qb  = b & 7;
    const int h   = (b >> 3) & 15;
    const int grp = b >> 7;                     // 0..6
    int dil, segBase;
    if (grp < 4)      { dil = 1; segBase = grp * 1024; }
    else if (grp < 6) { dil = 2; segBase = (grp - 4) * 2048; }
    else              { dil = 4; segBase = 0; }

    const int tid    = threadIdx.x;
    const int wave   = tid >> 5;
    const int lane   = tid & 31;
    const int laneHi = lane >> 4;
    const int ln     = lane & 15;
    const int col0   = h * HDIM;
    const int q0     = qb * 128;

    // ---- Q tile (128 x 64 bf16): async memory->LDS ----
    {
        int r  = tid >> 1;
        int ds = (tid & 1) * 32;
        const bf16* src = Pq + (size_t)(segBase + (q0 + r) * dil) * E_DIM + col0 + ds;
#pragma unroll
        for (int i = 0; i < 4; ++i)
            async_copy16(&sQ[r][ds + i * 8], src + i * 8);
    }

    const v8f vzero = {0.f,0.f,0.f,0.f,0.f,0.f,0.f,0.f};
    v8f o[4];
#pragma unroll
    for (int i = 0; i < 4; ++i) o[i] = vzero;
    float m_run[8], l_run[8];
#pragma unroll
    for (int v = 0; v < 8; ++v) { m_run[v] = -__builtin_inff(); l_run[v] = 0.f; }

    for (int kt = 0; kt < 16; ++kt) {
        const int key0 = kt * 64;
        __syncthreads();     // previous iteration's LDS consumers are done
        {
            int kk = tid >> 2;
            int ds = (tid & 3) * 16;
            size_t g = (size_t)(segBase + (key0 + kk) * dil) * E_DIM + col0 + ds;
            // K tile: straight copy -> async memory->LDS (2 x b128 per lane)
            const bf16* ksrc = Pk + g;
            async_copy16(&sK[kk][ds],     ksrc);
            async_copy16(&sK[kk][ds + 8], ksrc + 8);
            // V tile: VGPR-staged, stored transposed [dim][key]
            B16 vreg[2];
            vreg[0] = ((const B16*)(Pv + g))[0];
            vreg[1] = ((const B16*)(Pv + g))[1];
            const bf16* vs = (const bf16*)vreg;
#pragma unroll
            for (int i = 0; i < 16; ++i) sV[ds + i][kk] = vs[i];
        }
        if (tid < 64) sM[tid] = mask[segBase + (key0 + tid) * dil];
        wait_async0();       // ASYNCcnt drained before the barrier
        __syncthreads();

        // ---- S = Q K^T : wave's 16 rows x 64 keys ----
        v8f sacc[4];
#pragma unroll
        for (int i = 0; i < 4; ++i) sacc[i] = vzero;
#pragma unroll
        for (int ks = 0; ks < 2; ++ks) {
            FragB a;
            {
                const bf16* qp = &sQ[wave * 16 + ln][ks * 32];
                int c = laneHi * 16;
                a.q[0] = *(const B16*)(qp + c);
                a.q[1] = *(const B16*)(qp + 8 + c);
            }
            FragB bfr[4];
#pragma unroll
            for (int ct = 0; ct < 4; ++ct) {
                const bf16* kp = &sK[ct * 16 + ln][ks * 32 + laneHi * 16];
                bfr[ct].q[0] = ((const B16*)kp)[0];
                bfr[ct].q[1] = ((const B16*)kp)[1];
            }
#pragma unroll
            for (int ct = 0; ct < 4; ++ct)
                sacc[ct] = wmma_bf16(a, bfr[ct], sacc[ct]);
        }
        FragF s[4];
#pragma unroll
        for (int ct = 0; ct < 4; ++ct) {
            s[ct].v = sacc[ct];
            float mk = sM[ct * 16 + ln];
#pragma unroll
            for (int v = 0; v < 8; ++v) s[ct].f[v] = s[ct].f[v] * 0.125f - mk;
        }

        // ---- streaming softmax (rows fully owned per 16-lane half) ----
        float corr[8];
#pragma unroll
        for (int v = 0; v < 8; ++v) {
            float t = fmaxf(fmaxf(s[0].f[v], s[1].f[v]), fmaxf(s[2].f[v], s[3].f[v]));
#pragma unroll
            for (int off = 1; off < 16; off <<= 1) t = fmaxf(t, __shfl_xor(t, off, 32));
            float mnew = fmaxf(m_run[v], t);
            float mden = (mnew == -__builtin_inff()) ? 0.f : mnew;  // nan_to_num guard
            corr[v] = __expf(m_run[v] - mden);
            m_run[v] = mnew;
            float rs = 0.f;
#pragma unroll
            for (int ct = 0; ct < 4; ++ct) {
                float p = __expf(s[ct].f[v] - mden);
                rs += p;
                sP[wave][v + 8 * laneHi][ct * 16 + ln] = (bf16)p;
            }
#pragma unroll
            for (int off = 1; off < 16; off <<= 1) rs += __shfl_xor(rs, off, 32);
            l_run[v] = l_run[v] * corr[v] + rs;
        }
#pragma unroll
        for (int ct = 0; ct < 4; ++ct) {
            FragF t; t.v = o[ct];
#pragma unroll
            for (int v = 0; v < 8; ++v) t.f[v] *= corr[v];
            o[ct] = t.v;
        }
        __syncthreads();   // sP visible (uniform across waves)

        // ---- O += P V : A = P (k = keys), B = V^T-in-LDS (contiguous) ----
#pragma unroll
        for (int ks = 0; ks < 2; ++ks) {
            FragB a;
            {
                const bf16* pp = &sP[wave][ln][ks * 32];
                int c = laneHi * 16;
                a.q[0] = *(const B16*)(pp + c);
                a.q[1] = *(const B16*)(pp + 8 + c);
            }
            FragB bfr[4];
#pragma unroll
            for (int ct = 0; ct < 4; ++ct) {
                const bf16* vp = &sV[ct * 16 + ln][ks * 32 + laneHi * 16];
                bfr[ct].q[0] = ((const B16*)vp)[0];
                bfr[ct].q[1] = ((const B16*)vp)[1];
            }
#pragma unroll
            for (int ct = 0; ct < 4; ++ct)
                o[ct] = wmma_bf16(a, bfr[ct], o[ct]);
        }
    }

    // ---- epilogue: out[pos] += (O/l)/3, strided scatter ----
    float linv[8];
#pragma unroll
    for (int v = 0; v < 8; ++v) linv[v] = (l_run[v] > 0.f) ? 1.0f / l_run[v] : 0.f;
    const float third = 1.0f / 3.0f;
#pragma unroll
    for (int ct = 0; ct < 4; ++ct) {
        FragF c; c.v = o[ct];
#pragma unroll
        for (int v = 0; v < 8; ++v) {
            int qidx = q0 + wave * 16 + v + 8 * laneHi;
            size_t g = (size_t)(segBase + qidx * dil) * E_DIM + col0 + ct * 16 + ln;
            atomicAdd(out + g, c.f[v] * linv[v] * third);
        }
    }
}

// ===========================================================================
extern "C" void kernel_launch(void* const* d_in, const int* in_sizes, int n_in,
                              void* d_out, int out_size, void* d_ws, size_t ws_size,
                              hipStream_t stream) {
    (void)in_sizes; (void)n_in; (void)out_size; (void)ws_size;
    const float* q    = (const float*)d_in[0];
    const float* k    = (const float*)d_in[1];
    const float* v    = (const float*)d_in[2];
    const float* mask = (const float*)d_in[3];
    const float* Wq   = (const float*)d_in[4];
    float* out = (float*)d_out;
    bf16* Pbuf = (bf16*)d_ws;   // 3 * 4096 * 1024 bf16 = 25.2 MB scratch

    (void)hipMemsetAsync(d_out, 0, (size_t)S_LEN * E_DIM * sizeof(float), stream);
    dim3 pg(12288 / 128, E_DIM / 64);          // 96 x 16
    proj_kernel<<<pg, 256, 0, stream>>>(q, k, v, Wq, Pbuf);
    attn_kernel<<<896, 256, 0, stream>>>(Pbuf, mask, out);
}
